// SCON_78168404787848
// MI455X (gfx1250) — compile-verified, matrix operand order checked by reading, not run on hardware
//
#include <hip/hip_runtime.h>
#include <hip/hip_bf16.h>

typedef __attribute__((ext_vector_type(16))) _Float16 v16h;
typedef __attribute__((ext_vector_type(8)))  float    v8f;

// Pointer types matching the async-to-LDS builtin prototype (vV4i*1, V4i*3)
typedef int vint4_t __attribute__((vector_size(16)));
typedef __attribute__((address_space(1))) vint4_t* gv4p;   // global v4i*
typedef __attribute__((address_space(3))) vint4_t* lv4p;   // LDS v4i*

// Problem constants (from reference): T=200000, DT=0.1 -> N-1 = 2,000,000 steps
#define DTf 0.1f
constexpr int   OBS_EVERY = 100;
constexpr int   NCHUNK    = 20000;     // (N-1)/OBS_EVERY
constexpr int   NOBS      = 20001;
constexpr int   NGROUP    = 200;       // groups of chunks for the scan
constexpr int   CPG       = 100;       // chunks per group (NGROUP*CPG == NCHUNK)
constexpr float GAS_R     = 0.008314f;
constexpr float TEMP_REF  = 283.0f;
constexpr float TWO_PI    = 6.283185307179586f;

// Tile geometry for wave-cooperative noise staging in chunk_kernel
constexpr int TS = 20;                  // steps per tile (5 tiles of 20 = 100)
constexpr int SEG_F = TS * 3;           // 60 floats = 240 B per chunk segment (16B aligned)
constexpr int PIECES = 32 * (SEG_F/4);  // 480 b128 pieces per wave-tile (15 per lane)

// ---- probe CDNA5 async global->LDS path -------------------------------------
#if defined(__has_builtin)
#  if __has_builtin(__builtin_amdgcn_global_load_async_to_lds_b128)
#    define HAVE_ASYNC_LDS 1
#  else
#    define HAVE_ASYNC_LDS 0
#  endif
#  if __has_builtin(__builtin_amdgcn_s_wait_asynccnt)
#    define WAIT_ASYNC(n) __builtin_amdgcn_s_wait_asynccnt(n)
#  else
#    define WAIT_ASYNC(n) asm volatile("s_wait_asynccnt %0" :: "n"(n) : "memory")
#  endif
#else
#  define HAVE_ASYNC_LDS 0
#  define WAIT_ASYNC(n)
#endif

// ---- time-varying environment: k_S, k_D, k_M, I_S, I_D at time index ti ----
// Periods are exact in step units: daily = 24h/0.1h = 240, yearly = 8760h/0.1h = 87600
__device__ __forceinline__ void env_rates(int ti, const float* __restrict__ th,
                                          float& kS, float& kD, float& kM,
                                          float& IS, float& ID)
{
  int pd = ti % 240;
  int py = ti % 87600;
  float sd = sinf(TWO_PI * ((float)pd * (1.0f / 240.0f)));
  float sy = sinf(TWO_PI * ((float)py * (1.0f / 87600.0f)));
  float t  = (float)ti * DTf;
  float temp = TEMP_REF + 5.0f * (t * (1.0f / 700800.0f)) + 10.0f * sd + 10.0f * sy;
  float dinv = 1.0f / temp - 1.0f / TEMP_REF;
  kS = th[0] * expf(-th[3] * (1.0f / GAS_R) * dinv);
  kD = th[1] * expf(-th[4] * (1.0f / GAS_R) * dinv);
  kM = th[2] * expf(-th[5] * (1.0f / GAS_R) * dinv);
  IS = 0.001f  + 0.0005f  * sy;
  ID = 0.0001f + 0.00005f * sy;
}

// left-compose: (M,c) <- T o (M,c)   with T = [T.M | T.c] stored as 12 floats
__device__ __forceinline__ void compose_left(const float* __restrict__ T,
                                             float* __restrict__ M, float* __restrict__ c)
{
  float n00 = T[0]*M[0] + T[1]*M[3] + T[2]*M[6];
  float n01 = T[0]*M[1] + T[1]*M[4] + T[2]*M[7];
  float n02 = T[0]*M[2] + T[1]*M[5] + T[2]*M[8];
  float n10 = T[3]*M[0] + T[4]*M[3] + T[5]*M[6];
  float n11 = T[3]*M[1] + T[4]*M[4] + T[5]*M[7];
  float n12 = T[3]*M[2] + T[4]*M[5] + T[5]*M[8];
  float n20 = T[6]*M[0] + T[7]*M[3] + T[8]*M[6];
  float n21 = T[6]*M[1] + T[7]*M[4] + T[8]*M[7];
  float n22 = T[6]*M[2] + T[7]*M[5] + T[8]*M[8];
  float e0  = T[0]*c[0] + T[1]*c[1] + T[2]*c[2] + T[9];
  float e1  = T[3]*c[0] + T[4]*c[1] + T[5]*c[2] + T[10];
  float e2  = T[6]*c[0] + T[7]*c[1] + T[8]*c[2] + T[11];
  M[0]=n00; M[1]=n01; M[2]=n02; M[3]=n10; M[4]=n11; M[5]=n12;
  M[6]=n20; M[7]=n21; M[8]=n22; c[0]=e0; c[1]=e1; c[2]=e2;
}

// one Euler-Maruyama step folded into the running affine map (M,c)
__device__ __forceinline__ void step_compose(int ti, const float* __restrict__ th,
                                             float sc0, float sc1, float sc2,
                                             float n0, float n1, float n2,
                                             float* __restrict__ M, float* __restrict__ c)
{
  float kS, kD, kM, IS, ID;
  env_rates(ti, th, kS, kD, kM, IS, ID);

  // A = I + DT*W   (W row2 col0 == 0)
  float A00 = 1.0f - DTf * kS;
  float A01 = DTf * (th[6] * kD);                    // a_DS * k_D
  float A02 = DTf * (th[8] * th[9] * kM);            // a_M * a_MSC * k_M
  float A10 = DTf * (th[7] * kS);                    // a_SD * k_S
  float A11 = 1.0f - DTf * (th[10] + kD);            // -(u_M + k_D)
  float A12 = DTf * (th[8] * (1.0f - th[9]) * kM);
  float A21 = DTf * th[10];
  float A22 = 1.0f - DTf * kM;

  float d0 = DTf * IS + sc0 * n0;
  float d1 = DTf * ID + sc1 * n1;
  float d2 = sc2 * n2;

  float t00 = A00*M[0] + A01*M[3] + A02*M[6];
  float t01 = A00*M[1] + A01*M[4] + A02*M[7];
  float t02 = A00*M[2] + A01*M[5] + A02*M[8];
  float t10 = A10*M[0] + A11*M[3] + A12*M[6];
  float t11 = A10*M[1] + A11*M[4] + A12*M[7];
  float t12 = A10*M[2] + A11*M[5] + A12*M[8];
  float t20 =            A21*M[3] + A22*M[6];
  float t21 =            A21*M[4] + A22*M[7];
  float t22 =            A21*M[5] + A22*M[8];
  float e0  = A00*c[0] + A01*c[1] + A02*c[2] + d0;
  float e1  = A10*c[0] + A11*c[1] + A12*c[2] + d1;
  float e2  =            A21*c[1] + A22*c[2] + d2;
  M[0]=t00; M[1]=t01; M[2]=t02; M[3]=t10; M[4]=t11; M[5]=t12;
  M[6]=t20; M[7]=t21; M[8]=t22; c[0]=e0; c[1]=e1; c[2]=e2;
}

// ---------------- Stage 1: per-chunk (100-step) affine composition ----------------
// One thread per chunk; each wave owns 32 consecutive chunks and stages the noise
// for 20-step tiles through LDS with coalesced b128 transfers (async TDM-path when
// the toolchain exposes it, double-buffered so the DMA overlaps the VALU work).
__global__ void __launch_bounds__(128)
chunk_kernel(const float* __restrict__ theta, const float* __restrict__ noise,
             float* __restrict__ cms)
{
  const int tid  = threadIdx.x;
  const int lane = tid & 31;
  const int w    = tid >> 5;
  const int j    = blockIdx.x * 128 + tid;            // this thread's chunk
  const int waveChunkBase = blockIdx.x * 128 + w * 32;

  float th[11];
  #pragma unroll
  for (int i = 0; i < 11; ++i) th[i] = theta[i];
  float sc0 = sqrtf(fmaxf(theta[11], 1e-6f) * DTf);
  float sc1 = sqrtf(fmaxf(theta[12], 1e-6f) * DTf);
  float sc2 = sqrtf(fmaxf(theta[13], 1e-6f) * DTf);

  float M[9] = {1.f,0.f,0.f, 0.f,1.f,0.f, 0.f,0.f,1.f};
  float c[3] = {0.f,0.f,0.f};

#if HAVE_ASYNC_LDS
  __shared__ __align__(16) float stage[2][4][32][SEG_F];   // 61,440 B

  auto issue_tile = [&](int i0, int buf) {
    #pragma unroll
    for (int r = 0; r < PIECES / 32; ++r) {          // 15 b128 issues per lane
      int p    = r * 32 + lane;                      // piece id in [0,480)
      int cseg = p / 15;
      int q    = p - cseg * 15;
      int cidx = waveChunkBase + cseg;
      if (cidx > NCHUNK - 1) cidx = NCHUNK - 1;      // clamp: keep issue count uniform
      const float* src = noise + ((size_t)cidx * OBS_EVERY + i0) * 3 + q * 4;
      __builtin_amdgcn_global_load_async_to_lds_b128(
          (gv4p)src,
          (lv4p)&stage[buf][w][cseg][q * 4],
          0, 0);
    }
  };

  issue_tile(0, 0);
  for (int t = 0; t < OBS_EVERY / TS; ++t) {
    if (t < OBS_EVERY / TS - 1) {
      issue_tile((t + 1) * TS, (t + 1) & 1);         // prefetch next tile
      WAIT_ASYNC(15);                                // oldest 15 (tile t) complete
    } else {
      WAIT_ASYNC(0);
    }
    const float* S = &stage[t & 1][w][lane][0];
    #pragma unroll 4
    for (int i = 0; i < TS; ++i) {
      int ti = j * OBS_EVERY + t * TS + i + 1;       // weight/bias use time index s+1
      step_compose(ti, th, sc0, sc1, sc2,
                   S[i*3 + 0], S[i*3 + 1], S[i*3 + 2], M, c);
    }
  }
#else
  __shared__ __align__(16) float stage[4][32][SEG_F];      // 30,720 B
  for (int t = 0; t < OBS_EVERY / TS; ++t) {
    // wave-cooperative coalesced staging (same-wave DS ops are in-order)
    #pragma unroll
    for (int r = 0; r < PIECES / 32; ++r) {
      int p    = r * 32 + lane;
      int cseg = p / 15;
      int q    = p - cseg * 15;
      int cidx = waveChunkBase + cseg;
      if (cidx > NCHUNK - 1) cidx = NCHUNK - 1;
      const float4 v = *(const float4*)(noise + ((size_t)cidx * OBS_EVERY + t * TS) * 3 + q * 4);
      *(float4*)&stage[w][cseg][q * 4] = v;
    }
    const float* S = &stage[w][lane][0];
    #pragma unroll 4
    for (int i = 0; i < TS; ++i) {
      __builtin_prefetch(noise + ((size_t)j * OBS_EVERY + (t + 1) * TS) * 3, 0, 1);
      int ti = j * OBS_EVERY + t * TS + i + 1;
      step_compose(ti, th, sc0, sc1, sc2,
                   S[i*3 + 0], S[i*3 + 1], S[i*3 + 2], M, c);
    }
  }
#endif

  if (j < NCHUNK) {
    float* o = cms + (size_t)j * 12;
    #pragma unroll
    for (int r = 0; r < 9; ++r) o[r] = M[r];
    o[9] = c[0]; o[10] = c[1]; o[11] = c[2];
  }
}

// ---------------- Stage 2: hierarchical scan over chunk transforms ----------------
__global__ void __launch_bounds__(256)
scan_kernel(const float* __restrict__ x0, const float* __restrict__ cms,
            float* __restrict__ xobs)
{
  __shared__ float G[NGROUP][12];
  __shared__ float P[NGROUP][3];
  const int g = threadIdx.x;

  if (g < NGROUP) {   // phase 1: compose each group's 100 chunk transforms
    float M[9] = {1.f,0.f,0.f, 0.f,1.f,0.f, 0.f,0.f,1.f};
    float c[3] = {0.f,0.f,0.f};
    const float* T = cms + (size_t)g * CPG * 12;
    for (int k = 0; k < CPG; ++k, T += 12) compose_left(T, M, c);
    #pragma unroll
    for (int r = 0; r < 9; ++r) G[g][r] = M[r];
    G[g][9] = c[0]; G[g][10] = c[1]; G[g][11] = c[2];
  }
  __syncthreads();

  if (g == 0) {       // phase 2: short serial pivot over 200 group transforms
    float p0 = x0[0], p1 = x0[1], p2 = x0[2];
    for (int q = 0; q < NGROUP; ++q) {
      P[q][0] = p0; P[q][1] = p1; P[q][2] = p2;
      float q0 = G[q][0]*p0 + G[q][1]*p1 + G[q][2]*p2 + G[q][9];
      float q1 = G[q][3]*p0 + G[q][4]*p1 + G[q][5]*p2 + G[q][10];
      float q2 = G[q][6]*p0 + G[q][7]*p1 + G[q][8]*p2 + G[q][11];
      p0 = q0; p1 = q1; p2 = q2;
    }
    xobs[0] = x0[0]; xobs[1] = x0[1]; xobs[2] = x0[2];  // obs 0 = x0
  }
  __syncthreads();

  if (g < NGROUP) {   // phase 3: re-walk chunks, emit every observation state
    float p0 = P[g][0], p1 = P[g][1], p2 = P[g][2];
    const float* T = cms + (size_t)g * CPG * 12;
    float* xo = xobs + ((size_t)g * CPG + 1) * 3;
    for (int k = 0; k < CPG; ++k, T += 12, xo += 3) {
      float q0 = T[0]*p0 + T[1]*p1 + T[2]*p2 + T[9];
      float q1 = T[3]*p0 + T[4]*p1 + T[5]*p2 + T[10];
      float q2 = T[6]*p0 + T[7]*p1 + T[8]*p2 + T[11];
      p0 = q0; p1 = q1; p2 = q2;
      xo[0] = p0; xo[1] = p1; xo[2] = p2;
    }
  }
}

// ---------------- Stage 3: observation projection (WMMA batch of 16 dots) ----------------
// Each wave owns 16 observations (lanes 0..15). loc_y[n][0..2] = x; loc_y[n][3] = C1_n . x_n.
// The 16 length-3 dots are computed as the diagonal of a 16x16 v_wmma_f32_16x16x32_f16.
__global__ void __launch_bounds__(256)
obs_kernel(const float* __restrict__ theta, const float* __restrict__ xobs,
           float* __restrict__ out)
{
  const int lane  = threadIdx.x & 31;
  const int waveG = (blockIdx.x * blockDim.x + threadIdx.x) >> 5;
  const int m     = lane & 15;           // row/obs slot within the wave's tile
  const bool owner = (lane < 16);
  const int obs   = waveG * 16 + m;
  const bool valid = owner && (obs < NOBS);

  float c1x = 0.f, c1y = 0.f, c1z = 0.f;
  float xv0 = 0.f, xv1 = 0.f, xv2 = 0.f;
  if (valid) {
    float kS, kD, kM, IS, ID;
    env_rates(obs * OBS_EVERY, theta, kS, kD, kM, IS, ID);  // k at time index n*100
    c1x = (1.0f - theta[7]) * kS;   // (1 - a_SD) * ko_S
    c1y = (1.0f - theta[6]) * kD;   // (1 - a_DS) * ko_D
    c1z = (1.0f - theta[8]) * kM;   // (1 - a_M)  * ko_M
    xv0 = xobs[(size_t)obs * 3 + 0];
    xv1 = xobs[(size_t)obs * 3 + 1];
    xv2 = xobs[(size_t)obs * 3 + 2];
  }
  // EXEC reconverges here; WMMA executed by all lanes with operands zeroed
  // for non-owner / out-of-range lanes.

  // A (16x32 f16): row m = C1 of obs m in K=0..2.  Lanes 0-15 hold K=0..7,16..23.
  v16h a = {};
  a[0] = (_Float16)c1x; a[1] = (_Float16)c1y; a[2] = (_Float16)c1z;
  // B (32x16 f16): column n = x of obs n in K=0..2. Lanes 0-15 hold K=0..7,16..23.
  v16h b = {};
  b[0] = (_Float16)xv0; b[1] = (_Float16)xv1; b[2] = (_Float16)xv2;

  v8f cz = {};
  v8f d = __builtin_amdgcn_wmma_f32_16x16x32_f16(
      /*neg_a=*/false, a, /*neg_b=*/false, b,
      /*c_mod=*/(short)0, cz, /*reuse_a=*/false, /*reuse_b=*/false);

  // Diagonal element m lives at lane (m<8 ? m : m+16), VGPR m&7.
  int idx = lane & 7;
  float dv = d[0];
  dv = (idx == 1) ? d[1] : dv;
  dv = (idx == 2) ? d[2] : dv;
  dv = (idx == 3) ? d[3] : dv;
  dv = (idx == 4) ? d[4] : dv;
  dv = (idx == 5) ? d[5] : dv;
  dv = (idx == 6) ? d[6] : dv;
  dv = (idx == 7) ? d[7] : dv;
  int src = (m < 8) ? m : (m + 16);
  float comp3 = __shfl(dv, src, 32);

  if (valid) {
    float* o = out + (size_t)obs * 4;
    o[0] = xv0; o[1] = xv1; o[2] = xv2; o[3] = comp3;
  }
}

extern "C" void kernel_launch(void* const* d_in, const int* in_sizes, int n_in,
                              void* d_out, int out_size, void* d_ws, size_t ws_size,
                              hipStream_t stream)
{
  (void)in_sizes; (void)n_in; (void)out_size; (void)ws_size;
  const float* theta = (const float*)d_in[0];   // 14
  const float* x0    = (const float*)d_in[1];   // 3
  const float* noise = (const float*)d_in[2];   // 2,000,000 x 3
  float* out = (float*)d_out;                   // 20001 x 4

  float* cms  = (float*)d_ws;                   // NCHUNK*12 chunk transforms
  float* xobs = cms + (size_t)NCHUNK * 12;      // NOBS*3 observation states

  chunk_kernel<<<(NCHUNK + 127) / 128, 128, 0, stream>>>(theta, noise, cms);
  scan_kernel<<<1, 256, 0, stream>>>(x0, cms, xobs);

  // 16 obs per wave; 8 waves per 256-thread block -> 128 obs per block
  int obs_blocks = (NOBS + 127) / 128;
  obs_kernel<<<obs_blocks, 256, 0, stream>>>(theta, xobs, out);
}
